// ErniemoeMoE_47356309405781
// MI455X (gfx1250) — matrix operand order repaired and checked by reading, not compiled, not run on hardware
//
#include <hip/hip_runtime.h>
#include <hip/hip_bf16.h>

// ---------------------------------------------------------------------------
// ErnieMoE (T=2048, D=2048, E=16 top-4, I=1024, shared SI=2048) for gfx1250.
// bf16 WMMA (v_wmma_f32_16x16x32_bf16) GEMMs with fp32 accumulation.
// Double-buffered LDS, packed hw bf16 converts, SGPR-base + VGPR-offset
// addressing in the hot loop.
// ---------------------------------------------------------------------------

#define T_TOK 2048
#define DIM   2048
#define NEXP  16
#define TOPK  4
#define IEXP  1024
#define ISH   2048
#define NASSIGN (T_TOK * TOPK)   // 8192, fixed

typedef __attribute__((ext_vector_type(16))) __bf16 v16bf;
typedef __attribute__((ext_vector_type(2)))  __bf16 v2bf;
typedef __attribute__((ext_vector_type(8)))  float  v8f;
typedef __attribute__((ext_vector_type(2)))  float  v2f;

// Single RNE convert (backend picks hw op).
__device__ __forceinline__ unsigned short f2bf(float f) {
    __bf16 b = (__bf16)f;
    return __builtin_bit_cast(unsigned short, b);
}
// Packed RNE convert -> exactly one v_cvt_pk_bf16_f32.
__device__ __forceinline__ unsigned pack2bf(float lo, float hi) {
    v2f v = {lo, hi};
    v2bf b = __builtin_convertvector(v, v2bf);
    return __builtin_bit_cast(unsigned, b);
}
__device__ __forceinline__ float bf2f(unsigned short h) {
    unsigned u = (unsigned)h << 16;
    return __builtin_bit_cast(float, u);
}

// ---------------------------------------------------------------------------
// small helpers
// ---------------------------------------------------------------------------
__global__ void zero_counters_kernel(int* counts, int* cursors) {
    int i = threadIdx.x;
    if (i < NEXP) { counts[i] = 0; cursors[i] = 0; }
}

// 8 elements per thread, 2x float4 in, 1x uint4 (8 bf16) out
__global__ void convert_x_kernel(const float* __restrict__ x,
                                 unsigned short* __restrict__ xb, int n8) {
    int gid = blockIdx.x * blockDim.x + threadIdx.x;
    if (gid >= n8) return;
    const float4* src = (const float4*)(x) + gid * 2;
    float4 a = src[0], b = src[1];
    unsigned out[4];
    out[0] = pack2bf(a.x, a.y);
    out[1] = pack2bf(a.z, a.w);
    out[2] = pack2bf(b.x, b.y);
    out[3] = pack2bf(b.z, b.w);
    *((uint4*)(xb) + gid) = *(uint4*)out;
}

// ---------------------------------------------------------------------------
// Router: one wave per token. fp32 logits, softmax, bias-corrected top-4.
// ---------------------------------------------------------------------------
__global__ __launch_bounds__(256)
void router_kernel(const float* __restrict__ x, const float* __restrict__ gw,
                   const float* __restrict__ gbias,
                   int* __restrict__ top_idx, float* __restrict__ top_w,
                   int* __restrict__ counts) {
    const int wave = threadIdx.x >> 5;
    const int lane = threadIdx.x & 31;
    const int t = blockIdx.x * 8 + wave;
    if (t >= T_TOK) return;

    float acc[NEXP];
#pragma unroll
    for (int e = 0; e < NEXP; e++) acc[e] = 0.f;
    const float* xr = x + (size_t)t * DIM;
    for (int d = lane; d < DIM; d += 32) {
        float xv = xr[d];
#pragma unroll
        for (int e = 0; e < NEXP; e++) acc[e] += xv * gw[e * DIM + d];
    }
#pragma unroll
    for (int e = 0; e < NEXP; e++) {
#pragma unroll
        for (int off = 16; off > 0; off >>= 1)
            acc[e] += __shfl_xor(acc[e], off, 32);
    }
    if (lane == 0) {
        float mx = acc[0];
        for (int e = 1; e < NEXP; e++) mx = fmaxf(mx, acc[e]);
        float pr[NEXP]; float s = 0.f;
        for (int e = 0; e < NEXP; e++) { pr[e] = __expf(acc[e] - mx); s += pr[e]; }
        float inv = 1.f / s;
        for (int e = 0; e < NEXP; e++) pr[e] *= inv;

        float bmin = gbias[0];
        for (int e = 1; e < NEXP; e++) bmin = fminf(bmin, gbias[e]);

        int sel[TOPK]; float sw = 0.f; unsigned used = 0;
        for (int k = 0; k < TOPK; k++) {
            int best = 0; float bv = -1e30f;
            for (int e = 0; e < NEXP; e++) {
                if (used & (1u << e)) continue;
                float v = pr[e] + (gbias[e] - bmin);
                if (v > bv) { bv = v; best = e; }   // ties -> lowest index, like jax
            }
            sel[k] = best; used |= 1u << best; sw += pr[best];
        }
        float invw = 1.f / sw;
        for (int k = 0; k < TOPK; k++) {
            top_idx[t * TOPK + k] = sel[k];
            top_w[t * TOPK + k]   = pr[sel[k]] * invw;
            atomicAdd(&counts[sel[k]], 1);
        }
    }
}

__global__ void prefix_kernel(const int* __restrict__ counts, int* __restrict__ offs) {
    if (threadIdx.x == 0 && blockIdx.x == 0) {
        int run = 0;
        for (int e = 0; e < NEXP; e++) { offs[e] = run; run += counts[e]; }
    }
}

__global__ void scatter_kernel(const int* __restrict__ top_idx,
                               const int* __restrict__ offs, int* cursors,
                               int* __restrict__ tok, int* __restrict__ pos_of) {
    int a = blockIdx.x * blockDim.x + threadIdx.x;
    if (a >= NASSIGN) return;
    int e = top_idx[a];
    int pos = offs[e] + atomicAdd(&cursors[e], 1);
    tok[pos] = a >> 2;          // token id
    pos_of[a] = pos;
}

// ---------------------------------------------------------------------------
// Generic tiled bf16 WMMA GEMM: C[seg+m, n] = A[row(m), :] * Bf32[e][:, n]
// BM=128, BN=128, BK=32, 256 threads = 8 waves (2x4), wave does 64x32 (4x2
// wmma 16x16x32 tiles). Double-buffered LDS, one barrier per K-step,
// uniform SGPR base + 32-bit per-thread offsets for global loads.
// ---------------------------------------------------------------------------
#define BM 128
#define BN 128
#define BK 32
#define LPAD 40   // LDS row stride (bf16 units), 80B = multiple of 16B

// A: two 16B chunks per thread. pA uniform, offsets per-thread.
__device__ __forceinline__ void load_A_tile(const unsigned short* __restrict__ pA,
                                            unsigned aoff0, unsigned aoff1,
                                            unsigned short (*__restrict__ As)[LPAD],
                                            int r0, int c0, int r1, int c1) {
    *(uint4*)&As[r0][c0] = *(const uint4*)(pA + aoff0);
    *(uint4*)&As[r1][c1] = *(const uint4*)(pA + aoff1);
}

// B: 32x128 fp32 tile, convert to bf16, store transposed Bs[n][k].
// Rows k,k+1 loaded as float4 along n; (k,k+1) bf16 pairs produced by
// packed converts land as contiguous u32 in the transposed layout.
__device__ __forceinline__ void load_B_tile(const float* __restrict__ pB,
                                            unsigned boff0, unsigned boff1,
                                            int ldb,
                                            unsigned short (*__restrict__ Bs)[LPAD],
                                            int kk, int q) {
#pragma unroll
    for (int i = 0; i < 2; i++) {
        const float* src = pB + (i ? boff1 : boff0);   // k-pairs kk and kk+8
        int kc = 2 * (kk + i * 8);
        float4 a0 = *(const float4*)src;
        float4 a1 = *(const float4*)(src + ldb);
        *(unsigned*)&Bs[4 * q + 0][kc] = pack2bf(a0.x, a1.x);
        *(unsigned*)&Bs[4 * q + 1][kc] = pack2bf(a0.y, a1.y);
        *(unsigned*)&Bs[4 * q + 2][kc] = pack2bf(a0.z, a1.z);
        *(unsigned*)&Bs[4 * q + 3][kc] = pack2bf(a0.w, a1.w);
    }
}

__device__ __forceinline__ void compute_step(const unsigned short (*__restrict__ As)[LPAD],
                                             const unsigned short (*__restrict__ Bs)[LPAD],
                                             v8f acc[4][2],
                                             int wm, int wn, int half, int l16) {
    // A fragments: 16x32 bf16 layout -> lane(m=l16, half), K chunks
    // [half*8 .. half*8+7] and [16+half*8 .. 16+half*8+7]
    v16bf af[4];
#pragma unroll
    for (int mi = 0; mi < 4; mi++) {
        int r = wm * 64 + mi * 16 + l16;
        uint4* pa = (uint4*)&af[mi];
        pa[0] = *(const uint4*)&As[r][half * 8];
        pa[1] = *(const uint4*)&As[r][16 + half * 8];
    }
    // B fragments: 32x16 bf16 layout -> lane(n=l16, half), K chunks
    // [half*8 .. half*8+7] and [16+half*8 .. 16+half*8+7]
    v16bf bfr[2];
#pragma unroll
    for (int nj = 0; nj < 2; nj++) {
        int c = wn * 32 + nj * 16 + l16;
        uint4* pb = (uint4*)&bfr[nj];
        pb[0] = *(const uint4*)&Bs[c][half * 8];
        pb[1] = *(const uint4*)&Bs[c][16 + half * 8];
    }
#pragma unroll
    for (int mi = 0; mi < 4; mi++)
#pragma unroll
        for (int nj = 0; nj < 2; nj++)
            acc[mi][nj] = __builtin_amdgcn_wmma_f32_16x16x32_bf16(
                false, af[mi], false, bfr[nj], (short)0, acc[mi][nj],
                false, false);
}

template<bool GATHER, bool OUTF32>
__global__ __launch_bounds__(256)
void gemm_tile_kernel(const unsigned short* __restrict__ A, int lda,
                      const float* __restrict__ B, long long bStride, int ldb,
                      void* __restrict__ C, int ldc,
                      int M, int N, int K,
                      const int* __restrict__ counts, const int* __restrict__ offs,
                      const int* __restrict__ tok) {
    __shared__ unsigned short As[2][BM][LPAD];
    __shared__ unsigned short Bs[2][BN][LPAD];

    const int e = blockIdx.z;
    int cnt, segOff;
    if (counts) { cnt = counts[e]; segOff = offs[e]; }
    else        { cnt = M;         segOff = 0; }

    const int mBase = blockIdx.y * BM;
    if (mBase >= cnt) return;
    const int nBase = blockIdx.x * BN;

    const int tid  = threadIdx.x;
    const int lane = tid & 31;
    const int wave = tid >> 5;
    const int wm   = wave >> 2;   // 0..1
    const int wn   = wave & 3;    // 0..3
    const int half = lane >> 4;   // 0/1
    const int l16  = lane & 15;

    v8f acc[4][2] = {};

    // --- per-thread staging coordinates (fixed for whole kernel) -----------
    // A: thread stages chunks u = 2*tid, 2*tid+1 (row u>>2, 8-elem chunk u&3)
    const int r0 = (tid * 2)     >> 2, c0 = ((tid * 2)     & 3) * 8;
    const int r1 = (tid * 2 + 1) >> 2, c1 = ((tid * 2 + 1) & 3) * 8;
    int mrow0 = mBase + r0; mrow0 = (mrow0 < cnt) ? mrow0 : 0;
    int mrow1 = mBase + r1; mrow1 = (mrow1 < cnt) ? mrow1 : 0;
    const int arow0 = GATHER ? tok[segOff + mrow0] : (segOff + mrow0);
    const int arow1 = GATHER ? tok[segOff + mrow1] : (segOff + mrow1);
    // 32-bit element offsets (A <= 33.5MB, B <= 67MB: fit easily)
    const unsigned aoff0 = (unsigned)arow0 * (unsigned)lda + (unsigned)c0;
    const unsigned aoff1 = (unsigned)arow1 * (unsigned)lda + (unsigned)c1;
    const unsigned short* pA = A;                 // uniform, += BK per K-step

    // B: thread stages k-pairs kk, kk+8 at n-quad q
    const int kk = tid >> 5;      // 0..7
    const int q  = tid & 31;      // 0..31
    const float* pB = B + (size_t)e * (size_t)bStride;   // uniform, += BK*ldb
    const unsigned boff0 = (unsigned)(2 * kk) * (unsigned)ldb
                         + (unsigned)nBase + (unsigned)(4 * q);
    const unsigned boff1 = boff0 + (unsigned)(16 * ldb);
    const size_t bAdv = (size_t)BK * ldb;

    const int NK = K >> 5;        // even for all uses (64 or 32)
    load_A_tile(pA, aoff0, aoff1, As[0], r0, c0, r1, c1);
    load_B_tile(pB, boff0, boff1, ldb, Bs[0], kk, q);
    __syncthreads();
    for (int kt = 0; kt < NK; kt += 2) {
        // prefetch next tile into buffer 1 while computing buffer 0
        load_A_tile(pA + BK, aoff0, aoff1, As[1], r0, c0, r1, c1);
        load_B_tile(pB + bAdv, boff0, boff1, ldb, Bs[1], kk, q);
        compute_step(As[0], Bs[0], acc, wm, wn, half, l16);
        __syncthreads();
        pA += 2 * BK;             // uniform scalar adds
        pB += 2 * bAdv;
        if (kt + 2 < NK) {
            load_A_tile(pA, aoff0, aoff1, As[0], r0, c0, r1, c1);
            load_B_tile(pB, boff0, boff1, ldb, Bs[0], kk, q);
        }
        compute_step(As[1], Bs[1], acc, wm, wn, half, l16);
        __syncthreads();
    }

    // C 16x16 f32 layout: m = r + 8*half, n = l16
#pragma unroll
    for (int mi = 0; mi < 4; mi++) {
#pragma unroll
        for (int nj = 0; nj < 2; nj++) {
#pragma unroll
            for (int r = 0; r < 8; r++) {
                int gm = mBase + wm * 64 + mi * 16 + half * 8 + r;
                int gc = nBase + wn * 32 + nj * 16 + l16;
                if (gm < cnt) {
                    size_t off = (size_t)(segOff + gm) * ldc + gc;
                    if constexpr (OUTF32) ((float*)C)[off] = acc[mi][nj][r];
                    else ((unsigned short*)C)[off] = f2bf(acc[mi][nj][r]);
                }
            }
        }
    }
}

// ---------------------------------------------------------------------------
// SwiGLU on bf16 pairs: H[p,i] = silu(GU[p,i]) * GU[p, I+i]
// gu2/h2 are u32 views (2 bf16 per element); I2h = I2/2, Ih = I/2.
// ---------------------------------------------------------------------------
__global__ void swiglu_kernel(const unsigned* __restrict__ gu2,
                              unsigned* __restrict__ h2,
                              int rows, int I2h) {
    int Ih = I2h >> 1;
    size_t gid = (size_t)blockIdx.x * blockDim.x + threadIdx.x;
    if (gid >= (size_t)rows * Ih) return;
    int p = (int)(gid / Ih);
    int i = (int)(gid % Ih);
    unsigned gp = gu2[(size_t)p * I2h + i];
    unsigned up = gu2[(size_t)p * I2h + Ih + i];
    float g0 = bf2f((unsigned short)gp), g1 = bf2f((unsigned short)(gp >> 16));
    float u0 = bf2f((unsigned short)up), u1 = bf2f((unsigned short)(up >> 16));
    float s0 = g0 / (1.f + __expf(-g0)) * u0;
    float s1 = g1 / (1.f + __expf(-g1)) * u1;
    h2[gid] = pack2bf(s0, s1);
}

// out[t,d] += sum_k top_w[t,k] * Y[pos_of[t,k], d]; 2 columns per thread.
__global__ void combine_kernel(float* __restrict__ out,
                               const unsigned* __restrict__ Y2,   // bf16 pairs
                               const int* __restrict__ pos_of,
                               const float* __restrict__ top_w) {
    int gid = blockIdx.x * blockDim.x + threadIdx.x;   // over T*D/2
    if (gid >= T_TOK * DIM / 2) return;
    int t  = gid / (DIM / 2);
    int d2 = gid % (DIM / 2);
    float2 s = *((float2*)out + gid);
#pragma unroll
    for (int k = 0; k < TOPK; k++) {
        int pos = pos_of[t * TOPK + k];
        float w = top_w[t * TOPK + k];
        unsigned y = Y2[(size_t)pos * (DIM / 2) + d2];
        s.x += w * bf2f((unsigned short)y);
        s.y += w * bf2f((unsigned short)(y >> 16));
    }
    *((float2*)out + gid) = s;
}

// ---------------------------------------------------------------------------
extern "C" void kernel_launch(void* const* d_in, const int* in_sizes, int n_in,
                              void* d_out, int out_size, void* d_ws, size_t ws_size,
                              hipStream_t stream) {
    const float* x    = (const float*)d_in[0];
    const float* gw   = (const float*)d_in[1];
    const float* gb   = (const float*)d_in[2];
    const float* wgu  = (const float*)d_in[3];   // [E, D, 2I]
    const float* wdn  = (const float*)d_in[4];   // [E, I, D]
    const float* wsgu = (const float*)d_in[5];   // [D, 2*SI]
    const float* wsd  = (const float*)d_in[6];   // [SI, D]
    float* out = (float*)d_out;

    // workspace carve-up (256B aligned)
    char* ws = (char*)d_ws;
    size_t o = 0;
    auto alloc = [&](size_t bytes) { size_t r = o; o = (o + bytes + 255) & ~(size_t)255; return r; };
    unsigned short* xb   = (unsigned short*)(ws + alloc((size_t)T_TOK * DIM * 2));
    unsigned short* GU   = (unsigned short*)(ws + alloc((size_t)NASSIGN * 2 * IEXP * 2)); // also Y
    unsigned short* H    = (unsigned short*)(ws + alloc((size_t)NASSIGN * IEXP * 2));
    unsigned short* GUs  = (unsigned short*)(ws + alloc((size_t)T_TOK * 2 * ISH * 2));
    unsigned short* Hs   = (unsigned short*)(ws + alloc((size_t)T_TOK * ISH * 2));
    int* top_idx = (int*)(ws + alloc(NASSIGN * 4));
    float* top_w = (float*)(ws + alloc(NASSIGN * 4));
    int* pos_of  = (int*)(ws + alloc(NASSIGN * 4));
    int* tok     = (int*)(ws + alloc(NASSIGN * 4));
    int* counts  = (int*)(ws + alloc(256));
    int* offs    = (int*)(ws + alloc(256));
    int* cursors = (int*)(ws + alloc(256));
    unsigned short* Y = GU;   // alias: GU dead after swiglu, reuse for Y

    // 1. routing
    zero_counters_kernel<<<1, 32, 0, stream>>>(counts, cursors);
    convert_x_kernel<<<(T_TOK * DIM / 8) / 256, 256, 0, stream>>>(x, xb, T_TOK * DIM / 8);
    router_kernel<<<T_TOK / 8, 256, 0, stream>>>(x, gw, gb, top_idx, top_w, counts);
    prefix_kernel<<<1, 1, 0, stream>>>(counts, offs);
    scatter_kernel<<<NASSIGN / 256, 256, 0, stream>>>(top_idx, offs, cursors, tok, pos_of);

    // 2. routed experts: GU = gather(xb) @ w_gu[e]; H = swiglu(GU); Y = H @ w_dn[e]
    gemm_tile_kernel<true, false><<<dim3((2 * IEXP) / BN, T_TOK / BM, NEXP), 256, 0, stream>>>(
        xb, DIM, wgu, (long long)DIM * 2 * IEXP, 2 * IEXP,
        GU, 2 * IEXP, 0, 2 * IEXP, DIM, counts, offs, tok);
    swiglu_kernel<<<(NASSIGN * IEXP / 2) / 256, 256, 0, stream>>>(
        (const unsigned*)GU, (unsigned*)H, NASSIGN, IEXP);
    gemm_tile_kernel<false, false><<<dim3(DIM / BN, T_TOK / BM, NEXP), 256, 0, stream>>>(
        H, IEXP, wdn, (long long)IEXP * DIM, DIM,
        Y, DIM, 0, DIM, IEXP, counts, offs, nullptr);

    // 3. shared expert -> writes fp32 directly into d_out
    gemm_tile_kernel<false, false><<<dim3((2 * ISH) / BN, T_TOK / BM, 1), 256, 0, stream>>>(
        xb, DIM, wsgu, 0, 2 * ISH,
        GUs, 2 * ISH, T_TOK, 2 * ISH, DIM, nullptr, nullptr, nullptr);
    swiglu_kernel<<<(T_TOK * ISH / 2) / 256, 256, 0, stream>>>(
        (const unsigned*)GUs, (unsigned*)Hs, T_TOK, ISH);
    gemm_tile_kernel<false, true><<<dim3(DIM / BN, T_TOK / BM, 1), 256, 0, stream>>>(
        Hs, ISH, wsd, 0, DIM,
        out, DIM, T_TOK, DIM, ISH, nullptr, nullptr, nullptr);

    // 4. combine routed contributions into out
    combine_kernel<<<(T_TOK * DIM / 2) / 256, 256, 0, stream>>>(
        out, (const unsigned*)Y, pos_of, top_w);
}
